// ProdLSTMV1_56057913147658
// MI455X (gfx1250) — matrix operand order repaired and checked by reading, not compiled, not run on hardware
//
#include <hip/hip_runtime.h>
#include <hip/hip_bf16.h>

// ---------------------------------------------------------------------------
// ProdLSTM for MI455X (gfx1250, wave32, WMMA).
//
//  K1 build_x : gather/concat features -> bf16 X [B*L, 256] in workspace
//  K2 conv_w  : fp32 -> bf16 copies of W_ih / W_hh, bias = b_ih + b_hh
//  K3 lstm_recurrent : per-16-rows workgroup loops t=0..99; per timestep one
//     [16,512]x[512,1024] bf16 WMMA GEMM (A = x_t|h_{t-1} staged in LDS,
//     B = weight rows streamed from the 192MB L2 each step), fp32 accum,
//     cell state c lives entirely in VGPRs, h fed back through LDS in bf16,
//     out[b,t] = h . w_final + b_final emitted directly.
//
//  Round-2: anti-LICM asm stopped the weight fragments being hoisted and
//  spilled to scratch.
//  Round-3: address_space(1) typed weight pointers keep the anti-LICM asm
//  from laundering them to generic (flat_load); loads stay global_load_b128.
//  Round-4: HIP's uint4 is a class type whose operator= cannot bind an
//  AS(1) reference -> use a builtin ext_vector u32x4 for all 128-bit moves.
//
// Workspace: 4096*100*256*2 (X) + 2*1024*256*2 (W) + 4KB ~= 211 MB.
// ---------------------------------------------------------------------------

#define B_   4096
#define L_   100
#define F_   26
#define H_   256
#define DIN_ 256
#define G_   1024   // 4*H

typedef __attribute__((ext_vector_type(16))) __bf16        v16bf;
typedef __attribute__((ext_vector_type(8)))  float         v8f;
typedef __attribute__((ext_vector_type(4)))  unsigned int  u32x4;

typedef const __attribute__((address_space(1))) __bf16 gbf16;
typedef const __attribute__((address_space(1))) u32x4  gu32x4;

union FragBF {
  u32x4  u[2];
  v16bf  v;
};

__device__ __forceinline__ float sigm(float x) {
  return 1.0f / (1.0f + __expf(-x));
}
__device__ __forceinline__ float tanh_fast(float x) {
  float e = __expf(-2.0f * fabsf(x));
  float r = (1.0f - e) / (1.0f + e);
  return copysignf(r, x);
}

// ---------------------------------------------------------------------------
// K1: one wave per (b,t) row; lane covers columns lane, lane+32, ...
// ---------------------------------------------------------------------------
__global__ __launch_bounds__(256) void build_x(
    const float* __restrict__ inputs,
    const int* __restrict__ users, const int* __restrict__ products,
    const int* __restrict__ aisles, const int* __restrict__ depts,
    const int* __restrict__ dows, const int* __restrict__ hours,
    const int* __restrict__ tzs,
    const float* __restrict__ user_emb, const float* __restrict__ prod_emb,
    const float* __restrict__ aisle_emb, const float* __restrict__ dept_emb,
    __bf16* __restrict__ Xbf)
{
  size_t gtid  = (size_t)blockIdx.x * 256 + threadIdx.x;
  size_t rowid = gtid >> 5;                 // b*L + t
  int    lane  = (int)(gtid & 31);
  if (rowid >= (size_t)B_ * L_) return;

  int b = (int)(rowid / L_);
  int dow = dows[rowid], hour = hours[rowid], tz = tzs[rowid];
  int u = users[b], pr = products[b], ai = aisles[b], de = depts[b];

  for (int k = lane; k < DIN_; k += 32) {
    float v;
    if      (k < 26)  v = inputs[rowid * F_ + k];
    else if (k < 33)  v = (k - 26 == dow)  ? 1.0f : 0.0f;
    else if (k < 57)  v = (k - 33 == hour) ? 1.0f : 0.0f;
    else if (k < 85)  v = (k - 57 == tz)   ? 1.0f : 0.0f;
    else if (k < 106) v = aisle_emb[ai * 21 + (k - 85)];
    else if (k < 156) v = user_emb[(size_t)u * 50 + (k - 106)];
    else if (k < 206) v = prod_emb[(size_t)pr * 50 + (k - 156)];
    else              v = dept_emb[de * 50 + (k - 206)];
    Xbf[rowid * DIN_ + k] = (__bf16)v;
  }
}

// ---------------------------------------------------------------------------
// K2: weight precision conversion + bias fold.
// ---------------------------------------------------------------------------
__global__ __launch_bounds__(256) void conv_w(
    const float* __restrict__ W_ih, const float* __restrict__ W_hh,
    const float* __restrict__ b_ih, const float* __restrict__ b_hh,
    __bf16* __restrict__ WihB, __bf16* __restrict__ WhhB,
    float* __restrict__ bias)
{
  int i = blockIdx.x * 256 + threadIdx.x;
  if (i < G_ * DIN_) {
    WihB[i] = (__bf16)W_ih[i];
    WhhB[i] = (__bf16)W_hh[i];
  }
  if (i < G_) bias[i] = b_ih[i] + b_hh[i];
}

// ---------------------------------------------------------------------------
// K3: recurrent kernel. 256 blocks x 256 threads (8 waves). Wave w owns
// hidden columns [32w, 32w+32) across all 4 gates -> cell update lane-local.
// ---------------------------------------------------------------------------
__global__ __launch_bounds__(256) void lstm_recurrent(
    const __bf16* __restrict__ Xbf,
    const __bf16* __restrict__ Wih, const __bf16* __restrict__ Whh,
    const float* __restrict__ bias,
    const float* __restrict__ Wfin, const float* __restrict__ bfin,
    float* __restrict__ out)
{
  __shared__ __align__(16) __bf16 ldsA[16][DIN_ + H_];  // [m][k]: x_t | h_{t-1}
  __shared__ float ldsWf[H_];
  __shared__ float ldsPart[16][16];

  const int tid   = threadIdx.x;
  const int wave  = tid >> 5;
  const int lane  = tid & 31;
  const int row   = lane & 15;   // A-operand M row / B-operand N column
  const int khalf = lane >> 4;   // K-half selector per WMMA 16-bit layout
  const int b0    = blockIdx.x * 16;
  const int xm    = tid >> 4;    // staging: row 0..15
  const int xseg  = tid & 15;    // staging: 16-col segment

  if (tid < H_) ldsWf[tid] = Wfin[tid];
  for (int i = tid; i < 16 * H_; i += 256)
    ldsA[i >> 8][DIN_ + (i & 255)] = (__bf16)0.0f;   // h_{-1} = 0

  const int gbase = wave * 32 + row;   // gate column base within a 256 block
  float biasReg[4][2];
  #pragma unroll
  for (int p = 0; p < 4; ++p)
    #pragma unroll
    for (int q = 0; q < 2; ++q)
      biasReg[p][q] = bias[p * H_ + gbase + q * 16];

  // Per-lane element offset of each weight fragment row (k=0 slab).
  int wOff[4][2];
  #pragma unroll
  for (int p = 0; p < 4; ++p)
    #pragma unroll
    for (int q = 0; q < 2; ++q)
      wOff[p][q] = (p * H_ + gbase + q * 16) * DIN_ + khalf * 8;

  v8f c[2];
  #pragma unroll
  for (int q = 0; q < 2; ++q)
    #pragma unroll
    for (int j = 0; j < 8; ++j) c[q][j] = 0.0f;

  const float bfinal = bfin[0];
  const __bf16* xsrc0 =
      Xbf + ((size_t)(b0 + xm)) * L_ * DIN_ + xseg * 16;  // + t*DIN_ per step

  gbf16* WihG = (gbf16*)Wih;   // address_space(1): stays global through asm
  gbf16* WhhG = (gbf16*)Whh;
  __syncthreads();

  #pragma unroll 1
  for (int t = 0; t < L_; ++t) {
    // Defeat LICM: weight pointers are redefined every iteration so their
    // fragment loads stay per-step global_load_b128 (L2 hits) instead of
    // being hoisted and spilled to scratch. AS(1) typing keeps them off
    // the flat path (LOADcnt only, DS pipe left to the A-fragments).
    gbf16* WihL = WihG;
    gbf16* WhhL = WhhG;
    asm volatile("" : "+s"(WihL), "+s"(WhhL));

    // ---- stage x_t rows into LDS (32 B per thread) + prefetch x_{t+1} ----
    {
      const u32x4* src = (const u32x4*)(xsrc0 + (size_t)t * DIN_);
      u32x4 a0 = src[0];
      u32x4 a1 = src[1];
      u32x4* dst = (u32x4*)&ldsA[xm][xseg * 16];
      dst[0] = a0;
      dst[1] = a1;
      if (t + 1 < L_)
        __builtin_prefetch(xsrc0 + (size_t)(t + 1) * DIN_, 0, 1);
    }
    __syncthreads();

    // ---- accumulators = bias ----
    v8f acc[4][2];
    #pragma unroll
    for (int p = 0; p < 4; ++p)
      #pragma unroll
      for (int q = 0; q < 2; ++q) {
        float bb = biasReg[p][q];
        v8f a = {bb, bb, bb, bb, bb, bb, bb, bb};
        acc[p][q] = a;
      }

    // One K=32 slab: A from LDS (x|h), B rows from W (row-major [gate][k]
    // is exactly the per-lane-column layout WMMA's B operand wants).
    auto kstep = [&](gbf16* wmat, int kb, int koff) {
      FragBF af;
      const __bf16* arow = &ldsA[row][kb + khalf * 8];
      af.u[0] = *(const u32x4*)(arow);
      af.u[1] = *(const u32x4*)(arow + 16);
      #pragma unroll
      for (int p = 0; p < 4; ++p)
        #pragma unroll
        for (int q = 0; q < 2; ++q) {
          gu32x4* wp = (gu32x4*)(wmat + wOff[p][q] + koff);
          FragBF bf_;
          bf_.u[0] = wp[0];
          bf_.u[1] = wp[2];
          acc[p][q] = __builtin_amdgcn_wmma_f32_16x16x32_bf16(
              false, af.v, false, bf_.v, (short)0, acc[p][q], false, false);
        }
    };

    #pragma unroll
    for (int kt = 0; kt < 8; ++kt) kstep(WihL, kt * 32, kt * 32);        // x part
    #pragma unroll
    for (int kt = 0; kt < 8; ++kt) kstep(WhhL, DIN_ + kt * 32, kt * 32); // h part

    // ---- LSTM cell update (lane-local; c lives in VGPRs) ----
    float hval[2][8];
    #pragma unroll
    for (int q = 0; q < 2; ++q)
      #pragma unroll
      for (int j = 0; j < 8; ++j) {
        float ig = sigm(acc[0][q][j]);
        float fg = sigm(acc[1][q][j]);
        float gg = tanh_fast(acc[2][q][j]);
        float og = sigm(acc[3][q][j]);
        float cn = fg * c[q][j] + ig * gg;
        c[q][j] = cn;
        hval[q][j] = og * tanh_fast(cn);
      }

    __syncthreads();   // all WMMA LDS reads of this step done

    // ---- write h (bf16) back to LDS: C-tile row m = khalf*8 + j ----
    #pragma unroll
    for (int q = 0; q < 2; ++q)
      #pragma unroll
      for (int j = 0; j < 8; ++j)
        ldsA[khalf * 8 + j][DIN_ + wave * 32 + q * 16 + row] = (__bf16)hval[q][j];
    __syncthreads();

    // ---- out[b,t] = h . w_final + b_final (16 threads/row tree) ----
    {
      float s = 0.0f;
      #pragma unroll
      for (int j = 0; j < 16; ++j)
        s += (float)ldsA[xm][DIN_ + xseg * 16 + j] * ldsWf[xseg * 16 + j];
      ldsPart[xm][xseg] = s;
    }
    __syncthreads();
    if (tid < 16) {
      float s = bfinal;
      #pragma unroll
      for (int j = 0; j < 16; ++j) s += ldsPart[tid][j];
      out[((size_t)(b0 + tid)) * L_ + t] = s;
    }
    __syncthreads();
  }
}

// ---------------------------------------------------------------------------
extern "C" void kernel_launch(void* const* d_in, const int* in_sizes, int n_in,
                              void* d_out, int out_size, void* d_ws, size_t ws_size,
                              hipStream_t stream) {
  const float* inputs    = (const float*)d_in[0];
  const int*   users     = (const int*)d_in[1];
  const int*   products  = (const int*)d_in[2];
  const int*   aisles    = (const int*)d_in[3];
  const int*   depts     = (const int*)d_in[4];
  const int*   dows      = (const int*)d_in[5];
  const int*   hours     = (const int*)d_in[6];
  const int*   tzs       = (const int*)d_in[7];
  const float* user_emb  = (const float*)d_in[8];
  const float* prod_emb  = (const float*)d_in[9];
  const float* aisle_emb = (const float*)d_in[10];
  const float* dept_emb  = (const float*)d_in[11];
  const float* W_ih      = (const float*)d_in[12];
  const float* W_hh      = (const float*)d_in[13];
  const float* b_ih      = (const float*)d_in[14];
  const float* b_hh      = (const float*)d_in[15];
  const float* W_final   = (const float*)d_in[16];
  const float* b_final   = (const float*)d_in[17];
  float* out = (float*)d_out;

  char* ws = (char*)d_ws;
  constexpr size_t xbf_bytes = (size_t)B_ * L_ * DIN_ * sizeof(__bf16);
  constexpr size_t w_bytes   = (size_t)G_ * DIN_ * sizeof(__bf16);
  __bf16* Xbf  = (__bf16*)(ws);
  __bf16* WihB = (__bf16*)(ws + xbf_bytes);
  __bf16* WhhB = (__bf16*)(ws + xbf_bytes + w_bytes);
  float*  bias = (float*)(ws + xbf_bytes + 2 * w_bytes);

  build_x<<<(B_ * L_) / 8, 256, 0, stream>>>(
      inputs, users, products, aisles, depts, dows, hours, tzs,
      user_emb, prod_emb, aisle_emb, dept_emb, Xbf);
  conv_w<<<(G_ * DIN_) / 256, 256, 0, stream>>>(
      W_ih, W_hh, b_ih, b_hh, WihB, WhhB, bias);
  lstm_recurrent<<<B_ / 16, 256, 0, stream>>>(
      Xbf, WihB, WhhB, bias, W_final, b_final, out);
}
